// LSTMAdcNoise_5102421148035
// MI455X (gfx1250) — compile-verified, matrix operand order checked by reading, not run on hardware
//
#include <hip/hip_runtime.h>
#include <hip/hip_bf16.h>
#include <cstdint>

typedef __attribute__((ext_vector_type(16))) __bf16 bf16x16;
typedef __attribute__((ext_vector_type(8)))  float  f32x8;

#define HS_    2016
#define IN_F   128
#define PROJ_  504
#define NGATE  8064          // 4*HS
#define KC     640           // IN + PROJ padded to /32
#define KPK    2016          // proj GEMM K  (= HS, 63*32)
#define NP     512           // PROJ padded to /64
#define BATCH  256
#define TSEQ   256
#define GINV   (1.0f/75.0f)  // 1/G_RATIO, G_RATIO = 150/2

__device__ __forceinline__ float clipf(float w) {
    return fminf(fmaxf(w, -2.0f), 2.0f);
}

// CDNA5 async global->LDS DMA (bypasses VGPRs, tracked by ASYNCcnt).
__device__ __forceinline__ void async_ld16(unsigned ldsOff, unsigned long long gaddr) {
    asm volatile("global_load_async_to_lds_b128 %0, %1, off"
                 :: "v"(ldsOff), "v"(gaddr) : "memory");
}
__device__ __forceinline__ void wait_async0() {
    asm volatile("s_wait_asynccnt 0x0" ::: "memory");
}

// ---------------------------------------------------------------------------
// ADC threshold ramp (faithful fp64 reproduction of the numpy _ramp)
// thr[0..30] = sigmoid thresholds, thr[32..62] = tanh thresholds
// ---------------------------------------------------------------------------
__device__ void ramp_dev(const double* v, float* out) {
    double dv[32], r[32], cell[32];
    double rmin = 1e300, dvmax = -1e300, cmax = -1e300;
    for (int i = 0; i < 32; ++i) {
        dv[i] = v[i + 1] - v[i];
        r[i]  = rint(dv[i] * 1000.0) / 1000.0;   // np.round(.,3), RNE
        rmin  = fmin(rmin, r[i]);
        dvmax = fmax(dvmax, dv[i]);
    }
    for (int i = 0; i < 32; ++i) { cell[i] = r[i] / rmin; cmax = fmax(cmax, cell[i]); }
    double dva[32];
    for (int i = 0; i < 32; ++i) dva[i] = (150.0 / cmax * cell[i]) * (dvmax / 150.0);
    double V0 = 0.0;
    for (int i = 0; i < 16; ++i) V0 -= dva[i];
    out[0] = (float)V0;
    double V = V0;
    for (int i = 1; i < 31; ++i) { V += dva[i - 1]; out[i] = (float)V; }
}

__global__ void thr_init_kernel(float* __restrict__ thr) {
    if (threadIdx.x != 0 || blockIdx.x != 0) return;
    double v[33];
    for (int i = 0; i < 33; ++i) {               // sigmoid ramp
        double t = (double)(i + 1) / 34.0;
        v[i] = log(t / (1.0 - t));
    }
    ramp_dev(v, thr);
    for (int i = 0; i < 33; ++i) {               // tanh ramp
        double t = (double)(i - 16) / 17.0;
        v[i] = 0.5 * log((1.0 + t) / (1.0 - t));
    }
    ramp_dev(v, thr + 32);
}

// ---------------------------------------------------------------------------
// Weight prep: Wc (N-major [NGATE][KC] bf16) = clip([W;U]) + delg/G_RATIO
// ---------------------------------------------------------------------------
__global__ void wc_prep_kernel(const float* __restrict__ W, const float* __restrict__ U,
                               const float* __restrict__ delg, __bf16* __restrict__ Wc) {
    int idx = blockIdx.x * 256 + threadIdx.x;    // NGATE*KC threads exactly
    int n = idx / KC, k = idx - n * KC;
    float v = 0.0f;
    if (k < IN_F)
        v = clipf(W[(size_t)k * NGATE + n]) + delg[(size_t)k * NGATE + n] * GINV;
    else if (k < IN_F + PROJ_)
        v = clipf(U[(size_t)(k - IN_F) * NGATE + n]) + delg[(size_t)k * NGATE + n] * GINV;
    Wc[(size_t)n * KC + k] = (__bf16)v;
}

__global__ void bmu_prep_kernel(const float* __restrict__ bias,
                                const float* __restrict__ delg, float* __restrict__ bmu) {
    int n = blockIdx.x * 256 + threadIdx.x;
    if (n < NGATE)
        bmu[n] = clipf(bias[n]) + delg[(size_t)(IN_F + PROJ_) * NGATE + n] * GINV;
}

// Pw (N-major [NP][KPK] bf16) = proj_w (no clip/noise on proj weights), zero pad rows
__global__ void pw_prep_kernel(const float* __restrict__ proj_w, __bf16* __restrict__ Pw) {
    int idx = blockIdx.x * 256 + threadIdx.x;    // NP*KPK threads exactly
    int n = idx / KPK, k = idx - n * KPK;
    Pw[idx] = (__bf16)(n < PROJ_ ? proj_w[(size_t)n * KPK + k] : 0.0f);
}

// xh (bf16 [BATCH][KC]) = [ x[:,t,:] | h_prev | 0-pad ]
__global__ void prep_xh_kernel(const float* __restrict__ x, const float* __restrict__ hprev,
                               __bf16* __restrict__ xh, int t) {
    int idx = blockIdx.x * 256 + threadIdx.x;    // BATCH*KC threads exactly
    int b = idx / KC, k = idx - b * KC;
    float v = 0.0f;
    if (k < IN_F)              v = x[((size_t)b * TSEQ + t) * IN_F + k];
    else if (k < IN_F + PROJ_) v = hprev[b * PROJ_ + (k - IN_F)];
    xh[idx] = (__bf16)v;
}

// ---------------------------------------------------------------------------
// bf16 WMMA GEMM with async-to-LDS double buffering.
// out[M x N(f32)] = A[M x K(bf16 row-major)] * B (B stored N-major: Bg[n][k]).
// Block tile 128 x BN, 8 waves (4x2); wave tile 32 x (BN/2) = 2 x NT WMMA tiles.
// ---------------------------------------------------------------------------
template <int BN>
__launch_bounds__(256)
__global__ void gemm_bf16_kernel(const __bf16* __restrict__ Ag,
                                 const __bf16* __restrict__ Bg,
                                 const float*  __restrict__ bias,
                                 float* __restrict__ out,
                                 float* __restrict__ out2,
                                 int lda, int ldb, int K,
                                 int nTilesN, int outPitch, int out2Pitch,
                                 int validN) {
    constexpr int BM = 128, BK = 32, LDAs = BK + 8;  // +8 bf16 pad
    constexpr int NT = BN / 32;                      // 16-wide n-tiles per wave
    constexpr int NB = BN / 64;                      // B staging chunks per thread
    __shared__ __bf16 As[2][BM * LDAs];
    __shared__ __bf16 Bs[2][BN * LDAs];

    const int tid  = threadIdx.x;
    const int lane = tid & 31, wave = tid >> 5;
    const int wm = wave & 3, wn = wave >> 2;          // 4x2 wave grid
    const int m0 = (blockIdx.x / nTilesN) * BM;
    const int n0 = (blockIdx.x % nTilesN) * BN;
    const int half = lane >> 4, r = lane & 15;

    // staging: 16B (8 bf16) chunks; A rows tid/4, tid/4+64; B rows tid/4 + j*64
    const int rowA = tid >> 2, ko = (tid & 3) * 8;
    unsigned ldsA[2][2], ldsB[2][NB];
    const __bf16* gA[2];
    const __bf16* gB[NB];
    #pragma unroll
    for (int b2 = 0; b2 < 2; ++b2) {
        ldsA[b2][0] = (unsigned)(uintptr_t)&As[b2][rowA * LDAs + ko];
        ldsA[b2][1] = (unsigned)(uintptr_t)&As[b2][(rowA + 64) * LDAs + ko];
        #pragma unroll
        for (int j = 0; j < NB; ++j)
            ldsB[b2][j] = (unsigned)(uintptr_t)&Bs[b2][(rowA + j * 64) * LDAs + ko];
    }
    gA[0] = Ag + (size_t)(m0 + rowA) * lda + ko;
    gA[1] = Ag + (size_t)(m0 + rowA + 64) * lda + ko;
    #pragma unroll
    for (int j = 0; j < NB; ++j)
        gB[j] = Bg + (size_t)(n0 + rowA + j * 64) * ldb + ko;

    auto stage = [&](int buf, int kk) {
        async_ld16(ldsA[buf][0], (unsigned long long)(uintptr_t)(gA[0] + kk));
        async_ld16(ldsA[buf][1], (unsigned long long)(uintptr_t)(gA[1] + kk));
        #pragma unroll
        for (int j = 0; j < NB; ++j)
            async_ld16(ldsB[buf][j], (unsigned long long)(uintptr_t)(gB[j] + kk));
    };

    f32x8 acc[2][NT] = {};
    union Frag { bf16x16 v; unsigned u[8]; };

    stage(0, 0);
    const int nIter = K / BK;
    for (int it = 0; it < nIter; ++it) {
        const int buf = it & 1;
        wait_async0();          // this wave's DMA into As/Bs[buf] done
        __syncthreads();        // ... for all waves in the block
        if (it + 1 < nIter) stage(buf ^ 1, (it + 1) * BK);  // overlap with WMMA

        // ---- fragment extraction per ISA 16-bit A/B lane layout ----
        Frag aF[2], bF[NT];
        #pragma unroll
        for (int mi = 0; mi < 2; ++mi) {
            const unsigned* rp = (const unsigned*)(&As[buf][(wm * 32 + mi * 16 + r) * LDAs]);
            #pragma unroll
            for (int t2 = 0; t2 < 4; ++t2) {
                aF[mi].u[t2]     = rp[half * 4 + t2];       // K = half*8 + {0..7}
                aF[mi].u[4 + t2] = rp[8 + half * 4 + t2];   // K = 16 + half*8 + {0..7}
            }
        }
        #pragma unroll
        for (int ni = 0; ni < NT; ++ni) {
            const unsigned* rp =
                (const unsigned*)(&Bs[buf][(wn * (NT * 16) + ni * 16 + r) * LDAs]);
            #pragma unroll
            for (int t2 = 0; t2 < 4; ++t2) {
                bF[ni].u[t2]     = rp[half * 4 + t2];
                bF[ni].u[4 + t2] = rp[8 + half * 4 + t2];
            }
        }
        #pragma unroll
        for (int mi = 0; mi < 2; ++mi)
            #pragma unroll
            for (int ni = 0; ni < NT; ++ni)
                acc[mi][ni] = __builtin_amdgcn_wmma_f32_16x16x32_bf16(
                    false, aF[mi].v, false, bF[ni].v,
                    (short)0, acc[mi][ni], false, false);
    }

    // ---- epilogue: C/D layout row = e + 8*half, col = lane&15 ----
    #pragma unroll
    for (int mi = 0; mi < 2; ++mi) {
        #pragma unroll
        for (int ni = 0; ni < NT; ++ni) {
            int col   = n0 + wn * (NT * 16) + ni * 16 + r;
            int rbase = m0 + wm * 32 + mi * 16 + half * 8;
            if (col < validN) {
                float bv = bias ? bias[col] : 0.0f;
                #pragma unroll
                for (int e = 0; e < 8; ++e) {
                    float vv = acc[mi][ni][e] + bv;
                    out[(size_t)(rbase + e) * outPitch + col] = vv;
                    if (out2) out2[(size_t)(rbase + e) * out2Pitch + col] = vv;
                }
            }
        }
    }
}

// ---------------------------------------------------------------------------
// Pointwise: ADC-quantized gates, cell update, z = o * tanh(c)  (bf16 out)
// ---------------------------------------------------------------------------
__global__ void pointwise_kernel(const float* __restrict__ gates,
                                 const float* __restrict__ thr,
                                 float* __restrict__ c,
                                 __bf16* __restrict__ z) {
    __shared__ float sThr[64];
    if (threadIdx.x < 64) sThr[threadIdx.x] = thr[threadIdx.x];
    __syncthreads();
    int idx = blockIdx.x * 256 + threadIdx.x;    // BATCH*HS threads exactly
    int b = idx / HS_, j = idx - b * HS_;
    const float* g = gates + (size_t)b * NGATE;
    float gi = g[j], gf = g[HS_ + j], gg = g[2 * HS_ + j], go = g[3 * HS_ + j];
    int ci = 0, cf = 0, co = 0, cg = 0;
    #pragma unroll
    for (int i = 0; i < 31; ++i) {
        float s = sThr[i];
        ci += (gi >= s); cf += (gf >= s); co += (go >= s);
        cg += (gg >= sThr[32 + i]);
    }
    const float inv34 = 1.0f / 34.0f;
    float it = ci * inv34, ft = cf * inv34, ot = co * inv34;
    float gt = cg * (2.0f * inv34) - 1.0f;
    float cn = ft * c[idx] + it * gt;
    c[idx] = cn;
    z[idx] = (__bf16)(ot * tanhf(cn));
}

// ---------------------------------------------------------------------------
extern "C" void kernel_launch(void* const* d_in, const int* in_sizes, int n_in,
                              void* d_out, int out_size, void* d_ws, size_t ws_size,
                              hipStream_t stream) {
    const float* x      = (const float*)d_in[0];
    const float* W      = (const float*)d_in[1];
    const float* U      = (const float*)d_in[2];
    const float* bias   = (const float*)d_in[3];
    const float* proj_w = (const float*)d_in[4];
    const float* proj_b = (const float*)d_in[5];
    const float* delg   = (const float*)d_in[6];

    char* p = (char*)d_ws;
    auto alloc = [&](size_t bytes) -> void* {
        void* r = (void*)p;
        p += (bytes + 255) & ~(size_t)255;
        return r;
    };
    __bf16* Wc    = (__bf16*)alloc((size_t)NGATE * KC  * sizeof(__bf16)); // 10.3 MB
    __bf16* Pw    = (__bf16*)alloc((size_t)NP    * KPK * sizeof(__bf16)); //  2.1 MB
    float*  gates = (float*) alloc((size_t)BATCH * NGATE * sizeof(float));//  8.3 MB
    __bf16* xh    = (__bf16*)alloc((size_t)BATCH * KC  * sizeof(__bf16));
    __bf16* z     = (__bf16*)alloc((size_t)BATCH * KPK * sizeof(__bf16));
    float*  hprev = (float*) alloc((size_t)BATCH * PROJ_ * sizeof(float));
    float*  c     = (float*) alloc((size_t)BATCH * HS_   * sizeof(float));
    float*  bmu   = (float*) alloc((size_t)NGATE * sizeof(float));
    float*  thr   = (float*) alloc(64 * sizeof(float));

    hipMemsetAsync(hprev, 0, (size_t)BATCH * PROJ_ * sizeof(float), stream);
    hipMemsetAsync(c,     0, (size_t)BATCH * HS_   * sizeof(float), stream);

    thr_init_kernel<<<1, 1, 0, stream>>>(thr);
    wc_prep_kernel<<<(NGATE * KC) / 256, 256, 0, stream>>>(W, U, delg, Wc);
    bmu_prep_kernel<<<(NGATE + 255) / 256, 256, 0, stream>>>(bias, delg, bmu);
    pw_prep_kernel<<<(NP * KPK) / 256, 256, 0, stream>>>(proj_w, Pw);

    float* hs = (float*)d_out;                    // [B][T][PROJ]
    const int gGates = (NGATE / 128) * (BATCH / 128);  // 126 blocks, BN=128
    const int gProj  = (NP    / 64)  * (BATCH / 128);  // 16 blocks,  BN=64

    for (int t = 0; t < TSEQ; ++t) {
        prep_xh_kernel<<<(BATCH * KC) / 256, 256, 0, stream>>>(x, hprev, xh, t);
        gemm_bf16_kernel<128><<<gGates, 256, 0, stream>>>(
            xh, Wc, bmu, gates, nullptr,
            KC, KC, KC, NGATE / 128, NGATE, 0, NGATE);
        pointwise_kernel<<<(BATCH * HS_) / 256, 256, 0, stream>>>(gates, thr, c, z);
        gemm_bf16_kernel<64><<<gProj, 256, 0, stream>>>(
            z, Pw, proj_b, hs + (size_t)t * PROJ_, hprev,
            KPK, KPK, KPK, NP / 64, TSEQ * PROJ_, PROJ_, PROJ_);
    }

    float* hout = hs + (size_t)BATCH * TSEQ * PROJ_;
    hipMemcpyAsync(hout, hprev, (size_t)BATCH * PROJ_ * sizeof(float),
                   hipMemcpyDeviceToDevice, stream);
    hipMemcpyAsync(hout + (size_t)BATCH * PROJ_, c, (size_t)BATCH * HS_ * sizeof(float),
                   hipMemcpyDeviceToDevice, stream);
}
